// DeepCoevolve_81784767250561
// MI455X (gfx1250) — matrix operand order repaired and checked by reading, not compiled, not run on hardware
//
#include <hip/hip_runtime.h>

#define E     128
#define BMAX  16
#define NTH   768          // 24 waves of 32
#define MTILES 56          // 896 rows / 16
#define KB    4            // 128 / 32

typedef __attribute__((ext_vector_type(16))) _Float16     v16h;
typedef __attribute__((ext_vector_type(8)))  float        v8f;
typedef __attribute__((ext_vector_type(4)))  unsigned int u32x4;
typedef __attribute__((ext_vector_type(4)))  float        f32x4;

union frag16 {
    v16h  h;
    u32x4 q[2];
    unsigned int u[8];
    _Float16 e[16];
};

union halfpack4 {
    _Float16 e[4];
    unsigned int u[2];
    unsigned long long d;
};

__device__ __forceinline__ v8f wmma_f16(const frag16& a, const frag16& b, v8f c) {
    return __builtin_amdgcn_wmma_f32_16x16x32_f16(false, a.h, false, b.h,
                                                  (short)0, c, false, false);
}

// A fragments are pre-swizzled in workspace: block of 32 lanes x 16 halves per
// (sel, mtile, kblock); each lane loads 16 consecutive halves (2 x b128).
__device__ __forceinline__ frag16 load_a(const _Float16* __restrict__ WA,
                                         int sel, int m, int kb, int lane) {
    const u32x4* p = (const u32x4*)(WA + (((sel * MTILES + m) * KB + kb) << 9) + lane * 16);
    frag16 a; a.q[0] = p[0]; a.q[1] = p[1]; return a;
}

// B fragment (K=32 x N=16) from LDS f16 X (stored [event][k] row-major):
// lane: n = lane&15, g = lane>>4; K = kb*32 + g*16 + vg*2 + h (16 contiguous halves)
__device__ __forceinline__ frag16 build_b(const _Float16* Xh, int kb, int lane) {
    int n = lane & 15, g = lane >> 4;
    const u32x4* p = (const u32x4*)(Xh + n * E + kb * 32 + g * 16);
    frag16 b; b.q[0] = p[0]; b.q[1] = p[1]; return b;
}

__device__ __forceinline__ float sigm(float x) { return 1.f / (1.f + expf(-x)); }

// ---------------------------------------------------------------- kernels ---

__global__ void copy_tables(const float* __restrict__ ue, const float* __restrict__ ie,
                            float* __restrict__ Uw, float* __restrict__ Vw,
                            int nU, int nI) {
    int stride = gridDim.x * blockDim.x;
    int t0 = blockIdx.x * blockDim.x + threadIdx.x;
    for (int i = t0; i < nU; i += stride) Uw[i] = ue[i];
    for (int i = t0; i < nI; i += stride) Vw[i] = ie[i];
}

// Pre-swizzle stacked weight matrices into WMMA A-fragment order (f32 -> f16).
// sel 0: Wv = [ugru_wi ; igru_wh ; t1_w[:,128:]]   (multiplies v)
// sel 1: Wu = [ugru_wh ; igru_wi ; t1_w[:,:128]]   (multiplies u)
__global__ void prep_weights(const float* __restrict__ ugru_wi, const float* __restrict__ ugru_wh,
                             const float* __restrict__ igru_wi, const float* __restrict__ igru_wh,
                             const float* __restrict__ t1_w, _Float16* __restrict__ WA) {
    int idx = blockIdx.x * blockDim.x + threadIdx.x;
    const int total = 2 * MTILES * KB * 32 * 16;
    if (idx >= total) return;
    int i = idx & 15;  int t = idx >> 4;
    int lane = t & 31; t >>= 5;
    int kb = t & 3;    t >>= 2;
    int m = t % MTILES;
    int sel = t / MTILES;
    int g = lane >> 4;
    int mrow = m * 16 + (lane & 15);
    int vg = i >> 1, hh = i & 1;
    // A 16x32 layout (contiguous, matches B): K = g*16 + vg*2 + h
    int k = kb * 32 + g * 16 + vg * 2 + hh;
    float val;
    if (mrow < 384) {
        val = sel ? ugru_wh[mrow * E + k] : ugru_wi[mrow * E + k];
    } else if (mrow < 768) {
        int r = mrow - 384;
        val = sel ? igru_wi[r * E + k] : igru_wh[r * E + k];
    } else {
        int r = mrow - 768;
        val = t1_w[r * (2 * E) + (sel ? 0 : E) + k];
    }
    WA[((sel * MTILES + m) * KB + kb) * 512 + lane * 16 + i] = (_Float16)val;
}

// Greedy deterministic chunker: chunks of <=16 events with pairwise-distinct
// user ids AND item ids (such events are exactly independent in the scan).
__global__ void build_chunks(const int* __restrict__ uid, const int* __restrict__ iid,
                             int N, int* __restrict__ starts, int* __restrict__ ncnt) {
    if (blockIdx.x | threadIdx.x) return;
    int nc = 0, s = 0;
    while (s < N) {
        int e = s + 1;
        while (e < N && (e - s) < BMAX) {
            bool conf = false;
            for (int k = s; k < e; ++k)
                if (uid[k] == uid[e] || iid[k] == iid[e]) { conf = true; break; }
            if (conf) break;
            ++e;
        }
        starts[nc++] = s;
        s = e;
    }
    starts[nc] = N;
    *ncnt = nc;
}

__global__ __launch_bounds__(NTH, 1)
void coevolve_main(float* __restrict__ Uw, float* __restrict__ Vw,
                   const _Float16* __restrict__ WA,
                   const int* __restrict__ starts, const int* __restrict__ ncnt,
                   const int* __restrict__ user_ids, const int* __restrict__ item_ids,
                   const float* __restrict__ ubi, const float* __restrict__ ubh,
                   const float* __restrict__ ibi, const float* __restrict__ ibh,
                   const float* __restrict__ t1b, const float* __restrict__ t2b,
                   const float* __restrict__ t2w, const float* __restrict__ t3w,
                   const float* __restrict__ t3b, float* __restrict__ out) {
    __shared__ __align__(16) float Xu[BMAX * E], Xv[BMAX * E];
    __shared__ __align__(16) _Float16 Xuh[BMAX * E], Xvh[BMAX * E];
    __shared__ float H1[BMAX * E];
    __shared__ float H2[BMAX * 32];
    __shared__ float sUbi[384], sUbh[384], sIbi[384], sIbh[384];
    __shared__ float sT1b[E], sT2b[32], sT3w[32], sT3b;
    __shared__ float sT2w[32 * E];
    __shared__ int uids[BMAX], vids[BMAX];

    const int tid  = threadIdx.x;
    const int wave = tid >> 5;
    const int lane = tid & 31;

    for (int i = tid; i < 384; i += NTH) { sUbi[i]=ubi[i]; sUbh[i]=ubh[i]; sIbi[i]=ibi[i]; sIbh[i]=ibh[i]; }
    for (int i = tid; i < E;   i += NTH) sT1b[i] = t1b[i];
    for (int i = tid; i < 32*E;i += NTH) sT2w[i] = t2w[i];
    if (tid < 32) { sT2b[tid] = t2b[tid]; sT3w[tid] = t3w[tid]; }
    if (tid == 0) sT3b = t3b[0];
    const int nchunks = ncnt[0];

    // --- Hoist loop-invariant weight A-fragments into registers ------------
    // GRU waves (0..15): the 12 Wv fragments (tiles t0,t0+8,t0+16 x 4 kblocks);
    //   Wu fragments stay as in-loop loads (keeps VGPR use spill-free).
    // t1 waves (16..23): all 8 fragments (Wv m, Wu m x 4 kblocks).
    const bool isU = wave < 8;
    const int  g   = isU ? wave : wave - 8;
    const int  t0  = (isU ? 0 : 24) + g;
    frag16 aReg[12];
    if (wave < 16) {
        #pragma unroll
        for (int kb = 0; kb < KB; ++kb) {
            aReg[kb * 3 + 0] = load_a(WA, 0, t0,      kb, lane);
            aReg[kb * 3 + 1] = load_a(WA, 0, t0 + 8,  kb, lane);
            aReg[kb * 3 + 2] = load_a(WA, 0, t0 + 16, kb, lane);
        }
    } else {
        const int m = 48 + (wave - 16);
        #pragma unroll
        for (int kb = 0; kb < KB; ++kb) {
            aReg[kb * 3 + 0] = load_a(WA, 0, m, kb, lane);
            aReg[kb * 3 + 1] = load_a(WA, 1, m, kb, lane);
        }
    }
    __syncthreads();

    for (int c = 0; c < nchunks; ++c) {
        const int start = starts[c];
        const int Bc    = starts[c + 1] - start;

        if (tid < BMAX) {
            uids[tid] = (tid < Bc) ? user_ids[start + tid] : -1;
            vids[tid] = (tid < Bc) ? item_ids[start + tid] : -1;
        }
        __syncthreads();

        // Gather embedding rows into LDS, float4 per thread (rows contiguous).
        for (int idx = tid; idx < 2 * BMAX * (E / 4); idx += NTH) {
            int tab = idx >> 9;               // 0 = user, 1 = item
            int j   = (idx >> 5) & 15;        // event in chunk
            int q   = idx & 31;               // float4 within row
            int id  = tab ? vids[j] : uids[j];
            f32x4 val = {0.f, 0.f, 0.f, 0.f};
            if (id >= 0)
                val = *(const f32x4*)((tab ? Vw : Uw) + (size_t)id * E + q * 4);
            *(f32x4*)((tab ? Xv : Xu) + j * E + q * 4) = val;
            halfpack4 hp;
            hp.e[0] = (_Float16)val.x; hp.e[1] = (_Float16)val.y;
            hp.e[2] = (_Float16)val.z; hp.e[3] = (_Float16)val.w;
            *(unsigned long long*)((tab ? Xvh : Xuh) + j * E + q * 4) = hp.d;
        }
        __syncthreads();

        // Prefetch next chunk's rows (overlaps with WMMA below).
        if (c + 1 < nchunks && tid < 128) {
            int s2 = starts[c + 1];
            int B2 = starts[c + 2] - s2;
            int j = (tid >> 2) & 15, seg = (tid & 3) * 32;
            if (j < B2) {
                if (tid < 64) __builtin_prefetch(&Uw[(size_t)user_ids[s2 + j] * E + seg], 0, 0);
                else          __builtin_prefetch(&Vw[(size_t)item_ids[s2 + j] * E + seg], 0, 0);
            }
        }

        const int col = lane & 15;            // event within chunk (C-tile N)
        const int sub = (lane >> 4) * 8;      // C-tile M sub-offset

        if (wave < 16) {
            // --- coupled GRU: wave owns r/z/n tiles of both Yu and Yv ---
            v8f cv0 = {}, cv1 = {}, cv2 = {}, cu0 = {}, cu1 = {}, cu2 = {};
            #pragma unroll
            for (int kb = 0; kb < KB; ++kb) {
                frag16 bu = build_b(Xuh, kb, lane);
                frag16 bv = build_b(Xvh, kb, lane);
                cv0 = wmma_f16(aReg[kb * 3 + 0], bv, cv0);
                cv1 = wmma_f16(aReg[kb * 3 + 1], bv, cv1);
                cv2 = wmma_f16(aReg[kb * 3 + 2], bv, cv2);
                frag16 a;
                a = load_a(WA, 1, t0,      kb, lane); cu0 = wmma_f16(a, bu, cu0);
                a = load_a(WA, 1, t0 + 8,  kb, lane); cu1 = wmma_f16(a, bu, cu1);
                a = load_a(WA, 1, t0 + 16, kb, lane); cu2 = wmma_f16(a, bu, cu2);
            }
            // user: gi from Wv@v (cv*), gh from Wu@u (cu*), h=u
            // item: gi from Wu@u (cu*), gh from Wv@v (cv*), h=v
            const float* bi = isU ? sUbi : sIbi;
            const float* bh = isU ? sUbh : sIbh;
            const float* Hs = isU ? Xu : Xv;
            float* Tab = isU ? Uw : Vw;
            int id = (isU ? uids : vids)[col];
            #pragma unroll
            for (int i = 0; i < 8; ++i) {
                int row = g * 16 + sub + i;                      // 0..127
                float gir = (isU ? cv0[i] : cu0[i]) + bi[row];
                float ghr = (isU ? cu0[i] : cv0[i]) + bh[row];
                float r   = sigm(gir + ghr);
                float giz = (isU ? cv1[i] : cu1[i]) + bi[128 + row];
                float ghz = (isU ? cu1[i] : cv1[i]) + bh[128 + row];
                float z   = sigm(giz + ghz);
                float gin = (isU ? cv2[i] : cu2[i]) + bi[256 + row];
                float ghn = (isU ? cu2[i] : cv2[i]) + bh[256 + row];
                float n   = tanhf(gin + r * ghn);
                float hv  = Hs[col * E + row];
                float res = (1.f - z) * n + z * hv;
                if (id >= 0) Tab[(size_t)id * E + row] = res;    // coevolution scatter
            }
        } else {
            // --- t1: h1 = relu(Wu@u + Wv@v + t1_b) ---
            const int gw = wave - 16;
            v8f cv = {}, cu = {};
            #pragma unroll
            for (int kb = 0; kb < KB; ++kb) {
                frag16 bu = build_b(Xuh, kb, lane);
                frag16 bv = build_b(Xvh, kb, lane);
                cv = wmma_f16(aReg[kb * 3 + 0], bv, cv);
                cu = wmma_f16(aReg[kb * 3 + 1], bu, cu);
            }
            #pragma unroll
            for (int i = 0; i < 8; ++i) {
                int row = gw * 16 + sub + i;                     // 0..127
                float s = cv[i] + cu[i] + sT1b[row];
                H1[col * E + row] = s > 0.f ? s : 0.f;
            }
        }
        __threadfence();      // make scatters visible to next chunk's gathers
        __syncthreads();

        // --- t2 (512 dot-128), t3 + loss ---
        if (tid < BMAX * 32) {
            int e = tid >> 5, r = tid & 31;
            float acc = sT2b[r];
            #pragma unroll 4
            for (int k = 0; k < E; ++k) acc += sT2w[r * E + k] * H1[e * E + k];
            H2[e * 32 + r] = acc > 0.f ? acc : 0.f;
        }
        __syncthreads();
        if (tid < BMAX) {
            int e = tid;
            float s = sT3b;
            #pragma unroll
            for (int r = 0; r < 32; ++r) s += sT3w[r] * H2[e * 32 + r];
            if (e < Bc) out[(size_t)(start + e) * 2 + 1] = sigm(s);
        } else if (tid < 2 * BMAX) {
            int e = tid - BMAX;
            float d = 0.f;
            for (int k = 0; k < E; ++k) d += Xu[e * E + k] * Xv[e * E + k];
            float sp = (d > 20.f) ? d : log1pf(expf(d));
            if (e < Bc) out[(size_t)(start + e) * 2 + 0] = -logf(sp + 1e-10f);
        }
        __syncthreads();
    }
}

// ------------------------------------------------------------------ host ---

extern "C" void kernel_launch(void* const* d_in, const int* in_sizes, int n_in,
                              void* d_out, int out_size, void* d_ws, size_t ws_size,
                              hipStream_t stream) {
    const float* user_emb = (const float*)d_in[0];
    const float* item_emb = (const float*)d_in[1];
    const float* ugru_wi  = (const float*)d_in[2];
    const float* ugru_wh  = (const float*)d_in[3];
    const float* ugru_bi  = (const float*)d_in[4];
    const float* ugru_bh  = (const float*)d_in[5];
    const float* igru_wi  = (const float*)d_in[6];
    const float* igru_wh  = (const float*)d_in[7];
    const float* igru_bi  = (const float*)d_in[8];
    const float* igru_bh  = (const float*)d_in[9];
    const float* t1_w     = (const float*)d_in[10];
    const float* t1_b     = (const float*)d_in[11];
    const float* t2_w     = (const float*)d_in[12];
    const float* t2_b     = (const float*)d_in[13];
    const float* t3_w     = (const float*)d_in[14];
    const float* t3_b     = (const float*)d_in[15];
    const int* user_ids   = (const int*)d_in[16];
    const int* item_ids   = (const int*)d_in[17];
    float* out = (float*)d_out;

    const int NU  = in_sizes[0] / E;
    const int NI  = in_sizes[1] / E;
    const int NEV = in_sizes[16];

    char* ws = (char*)d_ws;
    size_t offU = 0;
    size_t offV = offU + (size_t)NU * E * sizeof(float);
    size_t offA = (offV + (size_t)NI * E * sizeof(float) + 255) & ~(size_t)255;
    size_t offS = (offA + (size_t)2 * MTILES * KB * 512 * sizeof(_Float16) + 255) & ~(size_t)255;
    size_t offN = offS + (size_t)(NEV + 1) * sizeof(int);

    float*    Uw     = (float*)(ws + offU);
    float*    Vw     = (float*)(ws + offV);
    _Float16* WA     = (_Float16*)(ws + offA);
    int*      starts = (int*)(ws + offS);
    int*      ncnt   = (int*)(ws + offN);

    copy_tables<<<2048, 256, 0, stream>>>(user_emb, item_emb, Uw, Vw, NU * E, NI * E);

    const int prepTotal = 2 * MTILES * KB * 32 * 16;
    prep_weights<<<(prepTotal + 255) / 256, 256, 0, stream>>>(
        ugru_wi, ugru_wh, igru_wi, igru_wh, t1_w, WA);

    build_chunks<<<1, 1, 0, stream>>>(user_ids, item_ids, NEV, starts, ncnt);

    coevolve_main<<<1, NTH, 0, stream>>>(
        Uw, Vw, WA, starts, ncnt, user_ids, item_ids,
        ugru_bi, ugru_bh, igru_bi, igru_bh,
        t1_b, t2_b, t2_w, t3_w, t3_b, out);
}
